// MOE_4887672783478
// MI455X (gfx1250) — compile-verified, matrix operand order checked by reading, not run on hardware
//
#include <hip/hip_runtime.h>
#include <hip/hip_bf16.h>

// ---------------------------------------------------------------------------
// MoE (E=8, top-2, SwiGLU 1024->2048->1024) for gfx1250 / MI455X.
// Compute-bound (~206 GFLOP) -> v_wmma_f32_16x16x32_bf16 (wave32 WMMA),
// double-buffered GLOBAL_LOAD_ASYNC_TO_LDS_B128 staging (ASYNCcnt).
// ---------------------------------------------------------------------------

typedef __attribute__((ext_vector_type(16))) __bf16 v16bf;
typedef __attribute__((ext_vector_type(8)))  __bf16 v8bf;
typedef __attribute__((ext_vector_type(8)))  float  v8f;

union BF16x16 { v16bf v; v8bf h[2]; };

static constexpr int E_   = 8;
static constexpr int K_   = 2;
static constexpr int C_   = 1024;
static constexpr int H_   = 2048;
static constexpr int B_   = 8;
static constexpr int T_   = 1024;
static constexpr int CAP  = K_ * T_;        // 2048 per-(batch,expert) capacity
static constexpr int NG   = B_ * E_;        // 64 groups
static constexpr int TOTAL_ROWS = B_ * T_ * K_;  // 16384 (exact, routing-independent)

// workspace layout (bytes)
static constexpr size_t OFF_CNT   = 0;                                  // 64 * 4
static constexpr size_t OFF_GOFF  = 256;                                // 64 * 4
static constexpr size_t OFF_TOK   = 512;                                // 64*2048*4
static constexpr size_t OFF_PROB  = OFF_TOK  + (size_t)NG * CAP * 4;    // 64*2048*4
static constexpr size_t OFF_XBF   = OFF_PROB + (size_t)NG * CAP * 4;    // B*T*C bf16
static constexpr size_t OFF_WFC   = OFF_XBF  + (size_t)B_ * T_ * C_ * 2;
static constexpr size_t OFF_WGT   = OFF_WFC  + (size_t)E_ * C_ * H_ * 2;
static constexpr size_t OFF_WPJ   = OFF_WGT  + (size_t)E_ * C_ * H_ * 2;
static constexpr size_t OFF_HG    = OFF_WPJ  + (size_t)E_ * C_ * H_ * 2;
// end = OFF_HG + TOTAL_ROWS*H_*2  (~178 MB total)

__device__ __forceinline__ unsigned short f2bf(float f) {
    unsigned int u = __float_as_uint(f);
    u += 0x7FFFu + ((u >> 16) & 1u);            // round-to-nearest-even
    return (unsigned short)(u >> 16);
}

// LDS byte offset of a shared-memory pointer (generic addr low 32 bits
// are the LDS offset per the gfx12 generic-address lowering / ISA 10.2).
__device__ __forceinline__ unsigned lds_off(const void* p) {
    return (unsigned)(uintptr_t)p;
}

// GLOBAL_LOAD_ASYNC_TO_LDS_B128, GVS mode: saddr base + 32-bit lane offset.
// Per lane: LDS[ldsAddr] = MEM[base + voff] (16 bytes).  Tracked by ASYNCcnt.
__device__ __forceinline__ void async_b128(unsigned ldsAddr, unsigned voff,
                                           const void* base) {
    asm volatile("global_load_async_to_lds_b128 %0, %1, %2"
                 :: "v"(ldsAddr), "v"(voff), "s"(base) : "memory");
}

#define WAIT_ASYNC(n) asm volatile("s_wait_asynccnt " #n ::: "memory")

// --------------------------- x -> bf16 -------------------------------------
__global__ void moe_cvt_x(const float* __restrict__ x, unsigned short* __restrict__ xb) {
    size_t i = ((size_t)blockIdx.x * blockDim.x + threadIdx.x) * 4;
    float4 v = *(const float4*)(x + i);
    ushort4 s;
    s.x = f2bf(v.x); s.y = f2bf(v.y); s.z = f2bf(v.z); s.w = f2bf(v.w);
    *(ushort4*)(xb + i) = s;
}

// ------------------- weight transpose + convert to bf16 --------------------
// src: [E][R][Cc] fp32  ->  dst: [E][Cc][R] bf16   (B-tiles become [n][k])
__global__ void moe_transp(const float* __restrict__ src, unsigned short* __restrict__ dst,
                           int R, int Cc) {
    __shared__ unsigned short tile[32][33];
    int e  = blockIdx.z;
    int c0 = blockIdx.x * 32, r0 = blockIdx.y * 32;
    int tx = threadIdx.x & 31, ty = threadIdx.x >> 5;      // ty in 0..7
    const float* s = src + (size_t)e * R * Cc;
    unsigned short* d = dst + (size_t)e * R * Cc;
#pragma unroll
    for (int j = 0; j < 4; ++j) {
        int r = r0 + ty + j * 8;
        tile[ty + j * 8][tx] = f2bf(s[(size_t)r * Cc + c0 + tx]);
    }
    __syncthreads();
#pragma unroll
    for (int j = 0; j < 4; ++j) {
        int c = c0 + ty + j * 8;
        d[(size_t)c * R + r0 + tx] = tile[tx][ty + j * 8];
    }
}

// ------------------------------- router -------------------------------------
// one wave32 per token: 8 logits, top-2 (JAX tie-break: lowest index),
// softmax over the two, append (token, prob) to expert slot lists.
__global__ void moe_router(const float* __restrict__ x, const float* __restrict__ rw,
                           int* __restrict__ cnt, int* __restrict__ tok,
                           float* __restrict__ prob) {
    int wid  = threadIdx.x >> 5;
    int lane = threadIdx.x & 31;
    int tid  = blockIdx.x * 8 + wid;            // flat token id in [0, B*T)
    int b = tid / T_, t = tid % T_;
    const float* xr = x + (size_t)tid * C_;
    float acc[E_];
#pragma unroll
    for (int e = 0; e < E_; ++e) acc[e] = 0.f;
    for (int c = lane; c < C_; c += 32) {
        float xv = xr[c];
        const float4* w = (const float4*)(rw + (size_t)c * E_);
        float4 w0 = w[0], w1 = w[1];
        acc[0] += xv * w0.x; acc[1] += xv * w0.y;
        acc[2] += xv * w0.z; acc[3] += xv * w0.w;
        acc[4] += xv * w1.x; acc[5] += xv * w1.y;
        acc[6] += xv * w1.z; acc[7] += xv * w1.w;
    }
#pragma unroll
    for (int e = 0; e < E_; ++e)
#pragma unroll
        for (int off = 16; off > 0; off >>= 1)
            acc[e] += __shfl_xor(acc[e], off, 32);
    if (lane == 0) {
        int e1 = 0; float l1 = acc[0];
#pragma unroll
        for (int e = 1; e < E_; ++e) if (acc[e] > l1) { l1 = acc[e]; e1 = e; }
        int e2 = 0; float l2 = -3.402823e38f; bool got = false;
#pragma unroll
        for (int e = 0; e < E_; ++e)
            if (e != e1 && (!got || acc[e] > l2)) { l2 = acc[e]; e2 = e; got = true; }
        float ex = __expf(l2 - l1);             // <= 1, no overflow
        float p1 = 1.f / (1.f + ex);
        float p2 = 1.f - p1;
        int g1 = b * E_ + e1, g2 = b * E_ + e2;
        int s1 = atomicAdd(&cnt[g1], 1);
        tok[g1 * CAP + s1] = t; prob[g1 * CAP + s1] = p1;
        int s2 = atomicAdd(&cnt[g2], 1);
        tok[g2 * CAP + s2] = t; prob[g2 * CAP + s2] = p2;
    }
}

// ----------------------- 64-entry exclusive scan ----------------------------
__global__ void moe_scan(const int* __restrict__ cnt, int* __restrict__ goff) {
    if (threadIdx.x == 0) {
        int run = 0;
        for (int i = 0; i < NG; ++i) { goff[i] = run; run += cnt[i]; }
    }
}

// ------------------- FFN stage 1: fc & gate GEMMs + SwiGLU ------------------
// grid: (group, Mtile<=16, H/64).  block 256 = 8 waves as 4(M) x 2(N).
// tile 128(M) x 64(N) x 64(K); double-buffered async LDS staging.
// Rows beyond cnt[] carry garbage in the A tile: harmless, since output row m
// depends only on A row m and out-of-count rows are never stored.
__global__ __launch_bounds__(256) void moe_ffn1(
    const unsigned short* __restrict__ xb,     // [B][T][C] bf16
    const unsigned short* __restrict__ wfc,    // [E][H][C] bf16 (transposed)
    const unsigned short* __restrict__ wgt,    // [E][H][C] bf16 (transposed)
    const float* __restrict__ bfc, const float* __restrict__ bgt,  // (E,1,H)
    const int* __restrict__ cnt, const int* __restrict__ goff,
    const int* __restrict__ tok,
    unsigned short* __restrict__ hg)           // [TOTAL_ROWS][H] bf16
{
    int grp = blockIdx.x;
    int m0  = blockIdx.y * 128;
    int rows = cnt[grp];
    if (m0 >= rows) return;
    int b = grp / E_, e = grp % E_;
    int n0 = blockIdx.z * 64;

    __shared__ unsigned short As[2][128][72];
    __shared__ unsigned short Bf[2][64][72];
    __shared__ unsigned short Bg[2][64][72];

    int tidx = threadIdx.x;
    int arow = tidx >> 1, aseg = tidx & 1;       // A: 2 thr/row, 64B each
    int brow = tidx >> 2, bseg = tidx & 3;       // B: 4 thr/row, 32B each
    int tIdx = tok[grp * CAP + min(m0 + arow, rows - 1)];   // clamped gather
    // per-thread global byte offsets at kb=0
    unsigned aOff = (unsigned)((((size_t)b * T_ + tIdx) * C_) * 2) + aseg * 64;
    unsigned fOff = (unsigned)((((size_t)e * H_ + n0 + brow) * C_) * 2) + bseg * 32;
    // per-thread LDS byte offsets (buffer 0)
    unsigned aLds = lds_off(&As[0][arow][aseg * 32]);
    unsigned bfLds = lds_off(&Bf[0][brow][bseg * 16]);
    unsigned bgLds = lds_off(&Bg[0][brow][bseg * 16]);
    const unsigned aBufStride  = (unsigned)sizeof(As[0]);
    const unsigned bBufStride  = (unsigned)sizeof(Bf[0]);

    int wid = tidx >> 5, lane = tidx & 31;
    int wm = wid >> 1, wn = wid & 1;
    int lrow = lane & 15, lhalf = lane >> 4;

    v8f accH[2][2], accG[2][2];
#pragma unroll
    for (int im = 0; im < 2; ++im)
#pragma unroll
        for (int in = 0; in < 2; ++in)
#pragma unroll
            for (int i = 0; i < 8; ++i) { accH[im][in][i] = 0.f; accG[im][in][i] = 0.f; }

    auto stage = [&](int kb, int buf) {
        unsigned ab = aLds + buf * aBufStride;
        unsigned fb = bfLds + buf * bBufStride;
        unsigned gb = bgLds + buf * bBufStride;
        unsigned ka = aOff + kb * 2, kf = fOff + kb * 2;
#pragma unroll
        for (int j = 0; j < 4; ++j) async_b128(ab + j * 16, ka + j * 16, xb);
#pragma unroll
        for (int j = 0; j < 2; ++j) {
            async_b128(fb + j * 16, kf + j * 16, wfc);
            async_b128(gb + j * 16, kf + j * 16, wgt);
        }
    };

    constexpr int NK = C_ / 64;       // 16 K-blocks
    stage(0, 0);                      // prologue
    for (int it = 0; it < NK; ++it) {
        int cur = it & 1;
        if (it + 1 < NK) { stage((it + 1) * 64, cur ^ 1); WAIT_ASYNC(8); }
        else             { WAIT_ASYNC(0); }
        __syncthreads();
#pragma unroll
        for (int kk = 0; kk < 64; kk += 32) {
            BF16x16 a[2], vf[2], vg[2];
#pragma unroll
            for (int im = 0; im < 2; ++im) {
                int m = wm * 32 + im * 16 + lrow;
                a[im].h[0] = *(const v8bf*)&As[cur][m][kk + lhalf * 8];
                a[im].h[1] = *(const v8bf*)&As[cur][m][kk + 16 + lhalf * 8];
            }
#pragma unroll
            for (int in = 0; in < 2; ++in) {
                int n = wn * 32 + in * 16 + lrow;
                vf[in].h[0] = *(const v8bf*)&Bf[cur][n][kk + lhalf * 8];
                vf[in].h[1] = *(const v8bf*)&Bf[cur][n][kk + 16 + lhalf * 8];
                vg[in].h[0] = *(const v8bf*)&Bg[cur][n][kk + lhalf * 8];
                vg[in].h[1] = *(const v8bf*)&Bg[cur][n][kk + 16 + lhalf * 8];
            }
#pragma unroll
            for (int im = 0; im < 2; ++im)
#pragma unroll
                for (int in = 0; in < 2; ++in) {
                    accH[im][in] = __builtin_amdgcn_wmma_f32_16x16x32_bf16(
                        false, a[im].v, false, vf[in].v, (short)0, accH[im][in], false, false);
                    accG[im][in] = __builtin_amdgcn_wmma_f32_16x16x32_bf16(
                        false, a[im].v, false, vg[in].v, (short)0, accG[im][in], false, false);
                }
        }
        __syncthreads();              // buffer free for issue at it+2
    }

    // epilogue: hg = (h+bh) * silu(g+bg), write bf16 to compacted buffer
    int rowBase = goff[grp];
#pragma unroll
    for (int im = 0; im < 2; ++im)
#pragma unroll
        for (int in = 0; in < 2; ++in) {
            int n = n0 + wn * 32 + in * 16 + lrow;
            float bh = bfc[e * H_ + n];
            float bg = bgt[e * H_ + n];
#pragma unroll
            for (int i = 0; i < 8; ++i) {
                int m = wm * 32 + im * 16 + lhalf * 8 + i;
                if (m0 + m < rows) {
                    float h = accH[im][in][i] + bh;
                    float g = accG[im][in][i] + bg;
                    float s = g / (1.f + __expf(-g));       // silu
                    hg[((size_t)(rowBase + m0 + m)) * H_ + n] = f2bf(h * s);
                }
            }
        }
}

// ---------------- FFN stage 2: proj GEMM + weighted combine -----------------
// grid: (group, Mtile<=16, C/128).  block 256 = 8 waves as 2(M) x 4(N).
// tile 128(M) x 128(N) x 64(K); double-buffered async LDS staging.
__global__ __launch_bounds__(256) void moe_ffn2(
    const unsigned short* __restrict__ hg,     // [TOTAL_ROWS][H] bf16
    const unsigned short* __restrict__ wpj,    // [E][C][H] bf16 (transposed)
    const float* __restrict__ bpj,             // (E,1,C)
    const int* __restrict__ cnt, const int* __restrict__ goff,
    const int* __restrict__ tok, const float* __restrict__ prob,
    float* __restrict__ y)                     // (B,T,C) fp32, pre-zeroed
{
    int grp = blockIdx.x;
    int m0  = blockIdx.y * 128;
    int rows = cnt[grp];
    if (m0 >= rows) return;
    int b = grp / E_, e = grp % E_;
    int n0 = blockIdx.z * 128;
    int rowBase = goff[grp];

    __shared__ unsigned short As[2][128][72];
    __shared__ unsigned short Bs[2][128][72];

    int tidx = threadIdx.x;
    int srow = tidx >> 1, sseg = tidx & 1;       // 2 thr/row, 64B each
    int rclamp = min(m0 + srow, rows - 1);       // clamped A-row gather
    unsigned aOff = (unsigned)(((size_t)(rowBase + rclamp) * H_) * 2) + sseg * 64;
    unsigned bOff = (unsigned)((((size_t)e * C_ + n0 + srow) * H_) * 2) + sseg * 64;
    unsigned aLds = lds_off(&As[0][srow][sseg * 32]);
    unsigned bLds = lds_off(&Bs[0][srow][sseg * 32]);
    const unsigned bufStride = (unsigned)sizeof(As[0]);

    int wid = tidx >> 5, lane = tidx & 31;
    int wm = wid >> 2, wn = wid & 3;
    int lrow = lane & 15, lhalf = lane >> 4;

    v8f acc[4][2];
#pragma unroll
    for (int im = 0; im < 4; ++im)
#pragma unroll
        for (int in = 0; in < 2; ++in)
#pragma unroll
            for (int i = 0; i < 8; ++i) acc[im][in][i] = 0.f;

    auto stage = [&](int kb, int buf) {
        unsigned ab = aLds + buf * bufStride;
        unsigned bb = bLds + buf * bufStride;
        unsigned ka = aOff + kb * 2, kn = bOff + kb * 2;
#pragma unroll
        for (int j = 0; j < 4; ++j) {
            async_b128(ab + j * 16, ka + j * 16, hg);
            async_b128(bb + j * 16, kn + j * 16, wpj);
        }
    };

    constexpr int NK = H_ / 64;       // 32 K-blocks
    stage(0, 0);
    for (int it = 0; it < NK; ++it) {
        int cur = it & 1;
        if (it + 1 < NK) { stage((it + 1) * 64, cur ^ 1); WAIT_ASYNC(8); }
        else             { WAIT_ASYNC(0); }
        __syncthreads();
#pragma unroll
        for (int kk = 0; kk < 64; kk += 32) {
            BF16x16 a[4], bm[2];
#pragma unroll
            for (int im = 0; im < 4; ++im) {
                int m = wm * 64 + im * 16 + lrow;
                a[im].h[0] = *(const v8bf*)&As[cur][m][kk + lhalf * 8];
                a[im].h[1] = *(const v8bf*)&As[cur][m][kk + 16 + lhalf * 8];
            }
#pragma unroll
            for (int in = 0; in < 2; ++in) {
                int n = wn * 32 + in * 16 + lrow;
                bm[in].h[0] = *(const v8bf*)&Bs[cur][n][kk + lhalf * 8];
                bm[in].h[1] = *(const v8bf*)&Bs[cur][n][kk + 16 + lhalf * 8];
            }
#pragma unroll
            for (int im = 0; im < 4; ++im)
#pragma unroll
                for (int in = 0; in < 2; ++in)
                    acc[im][in] = __builtin_amdgcn_wmma_f32_16x16x32_bf16(
                        false, a[im].v, false, bm[in].v, (short)0, acc[im][in], false, false);
        }
        __syncthreads();
    }

    // epilogue: y[b,t,n] += p * (o + bias)   (2 terms/element -> deterministic)
#pragma unroll
    for (int im = 0; im < 4; ++im)
#pragma unroll
        for (int i = 0; i < 8; ++i) {
            int m = wm * 64 + im * 16 + lhalf * 8 + i;
            if (m0 + m < rows) {
                int   t = tok[grp * CAP + m0 + m];
                float p = prob[grp * CAP + m0 + m];
                float* yr = y + ((size_t)b * T_ + t) * C_;
#pragma unroll
                for (int in = 0; in < 2; ++in) {
                    int n = n0 + wn * 32 + in * 16 + lrow;
                    float v = acc[im][in][i] + bpj[e * C_ + n];
                    atomicAdd(&yr[n], p * v);
                }
            }
        }
}

// ----------------------------------------------------------------------------
extern "C" void kernel_launch(void* const* d_in, const int* in_sizes, int n_in,
                              void* d_out, int out_size, void* d_ws, size_t ws_size,
                              hipStream_t stream) {
    const float* x   = (const float*)d_in[0];
    const float* rw  = (const float*)d_in[1];
    const float* wfc = (const float*)d_in[2];
    const float* bfc = (const float*)d_in[3];
    const float* wgt = (const float*)d_in[4];
    const float* bgt = (const float*)d_in[5];
    const float* wpj = (const float*)d_in[6];
    const float* bpj = (const float*)d_in[7];
    float* y = (float*)d_out;
    char* ws = (char*)d_ws;

    int*   cnt  = (int*)(ws + OFF_CNT);
    int*   goff = (int*)(ws + OFF_GOFF);
    int*   tok  = (int*)(ws + OFF_TOK);
    float* prob = (float*)(ws + OFF_PROB);
    unsigned short* xb   = (unsigned short*)(ws + OFF_XBF);
    unsigned short* wfcT = (unsigned short*)(ws + OFF_WFC);
    unsigned short* wgtT = (unsigned short*)(ws + OFF_WGT);
    unsigned short* wpjT = (unsigned short*)(ws + OFF_WPJ);
    unsigned short* hg   = (unsigned short*)(ws + OFF_HG);

    hipMemsetAsync(y, 0, (size_t)out_size * sizeof(float), stream);
    hipMemsetAsync(cnt, 0, 512, stream);

    moe_cvt_x<<<(B_ * T_ * C_) / (256 * 4), 256, 0, stream>>>(x, xb);
    moe_transp<<<dim3(H_ / 32, C_ / 32, E_), 256, 0, stream>>>(wfc, wfcT, C_, H_);
    moe_transp<<<dim3(H_ / 32, C_ / 32, E_), 256, 0, stream>>>(wgt, wgtT, C_, H_);
    moe_transp<<<dim3(C_ / 32, H_ / 32, E_), 256, 0, stream>>>(wpj, wpjT, H_, C_);
    moe_router<<<(B_ * T_) / 8, 256, 0, stream>>>(x, rw, cnt, tok, prob);
    moe_scan<<<1, 32, 0, stream>>>(cnt, goff);
    moe_ffn1<<<dim3(NG, CAP / 128, H_ / 64), 256, 0, stream>>>(
        xb, wfcT, wgtT, bfc, bgt, cnt, goff, tok, hg);
    moe_ffn2<<<dim3(NG, CAP / 128, C_ / 128), 256, 0, stream>>>(
        hg, wpjT, bpj, cnt, goff, tok, prob, y);
}
